// IMDE_80564996538442
// MI455X (gfx1250) — matrix operand order repaired and checked by reading, not compiled, not run on hardware
//
#include <hip/hip_runtime.h>
#include <hip/hip_fp16.h>
#include <cstdint>
#include <cstddef>

// ---------------------------------------------------------------------------
// IMDE wavelet-fusion forward for MI455X (gfx1250, wave32, WMMA).
// - All token-level linear algebra (q/k/v/proj, MLP, 1x1 gate convs, 3x3 convs
//   as 9-tap implicit GEMM) runs on v_wmma_f32_16x16x32_f16, f32 accumulate.
// - Weights are pre-packed ONCE per launch into f16 B-fragment lane order in
//   global memory; each block then stages them into LDS with
//   global_load_async_to_lds_b128 (ASYNCcnt) -- pure async copy, no VALU.
// ---------------------------------------------------------------------------

typedef __attribute__((ext_vector_type(16))) _Float16 v16h;
typedef __attribute__((ext_vector_type(8)))  _Float16 v8h;
typedef __attribute__((ext_vector_type(8)))  float    v8f;

static constexpr int NTOK = 65536;   // B*128*128
static constexpr int HB   = 128;     // half spatial

__device__ __forceinline__ float geluf(float x) {
    return 0.5f * x * (1.0f + erff(x * 0.70710678118654752f));
}
__device__ __forceinline__ float siluf(float x) {
    return x / (1.0f + expf(-x));
}

// Async copy of `bytes` (multiple of 4096) from packed global weights to LDS.
// Each of 256 threads issues b128 async-to-LDS ops; completion via ASYNCcnt.
__device__ __forceinline__ void async_stage(const _Float16* __restrict__ src,
                                            _Float16* __restrict__ lds,
                                            int bytes, int tid)
{
    const unsigned ldsbase = (unsigned)(size_t)lds;  // low 32 bits = LDS offset
    const unsigned long long s64 = (unsigned long long)(size_t)src;
    for (int off = tid * 16; off < bytes; off += 4096) {
        unsigned la = ldsbase + (unsigned)off;
        asm volatile("global_load_async_to_lds_b128 %0, %1, %2"
                     : : "v"(la), "v"(off), "s"(s64) : "memory");
    }
    asm volatile("s_wait_asynccnt 0" : : : "memory");
}

// ---------------------------------------------------------------------------
// Weight pre-pack: f32 [NOUT,K] row-major -> f16 fragments, frag = kt*NT + nt,
// within-frag index = lane*16 + h, element (k = kt*32 + (lane>>4)*16 + h,
// n = nt*16 + (lane&15)). One thread per element; grid = K*NOUT/256.
// ---------------------------------------------------------------------------
__global__ void prepack_gemm(const float* __restrict__ W, _Float16* __restrict__ P,
                             int K, int NT)
{
    int idx  = blockIdx.x * 256 + threadIdx.x;
    int frag = idx >> 9, w = idx & 511, l = w >> 4, h = w & 15;
    int kt   = frag / NT;
    int nt   = frag - kt * NT;
    int k    = kt * 32 + ((l >> 4) << 4) + h;
    int n    = nt * 16 + (l & 15);
    P[idx] = (_Float16)W[(size_t)n * K + k];
}

// conv weight [O=64,I=64,3,3] -> f16 fragments, frag = tap*8 + kt*4 + nt
__global__ void prepack_conv(const float* __restrict__ W, _Float16* __restrict__ P)
{
    int idx  = blockIdx.x * 256 + threadIdx.x;   // 9*4096 elements
    int frag = idx >> 9, w = idx & 511, l = w >> 4, h = w & 15;
    int tap  = frag >> 3, kt = (frag >> 2) & 1, nt = frag & 3;
    int k    = kt * 32 + ((l >> 4) << 4) + h;
    int n    = nt * 16 + (l & 15);
    P[idx] = (_Float16)W[(size_t)n * 576 + k * 9 + tap];
}

// ---------------------------------------------------------------------------
// Generic token GEMM: Y[tok, NOUT] = epilogue( X[tok, K](f16) @ W^T + b )
// Wp = pre-packed f16 weight fragments.
// EP: 0 = store f32; 2 = gelu -> f16; 3 = silu -> f32; 4 = Res + y -> f32;
//     5 = Acc += Res + y
// grid: NTOK/128 blocks of 256 threads (8 waves, each wave one 16-token M tile)
// ---------------------------------------------------------------------------
template <int K, int NOUT, int EP>
__global__ __launch_bounds__(256) void gemm_tok(
    const _Float16* __restrict__ X, const _Float16* __restrict__ Wp,
    const float* __restrict__ bias, float* __restrict__ Of,
    _Float16* __restrict__ Oh, const float* __restrict__ Res,
    float* __restrict__ Acc)
{
    constexpr int KT = K / 32;
    constexpr int NT = NOUT / 16;
    // Chunk the kt dimension so static LDS stays <= ~36KB.
    constexpr int KTC    = (K * NOUT * 2 > 49152) ? (KT / 2) : KT;
    constexpr int CHUNKB = KTC * NT * 1024;      // bytes per staged chunk
    __shared__ __align__(64) _Float16 wl[KTC * NT * 512];

    const int wave = threadIdx.x >> 5;
    const int lane = threadIdx.x & 31;
    const int tok0 = ((blockIdx.x << 3) + wave) << 4;
    const int row  = tok0 + (lane & 15);
    const int kb   = lane >> 4;
    const _Float16* __restrict__ xr = X + (size_t)row * K;
    __builtin_prefetch(xr, 0, 0);

    v8f acc[NT];
#pragma unroll
    for (int i = 0; i < NT; i++)
#pragma unroll
        for (int j = 0; j < 8; j++) acc[i][j] = 0.0f;

    for (int kc = 0; kc < KT; kc += KTC) {
        __syncthreads();
        async_stage(Wp + (size_t)kc * NT * 512, wl, CHUNKB, threadIdx.x);
        __syncthreads();

#pragma unroll
        for (int kt2 = 0; kt2 < KTC; kt2++) {
            const int kt = kc + kt2;
            // A fragment (16x32 f16): lane row, halves 0..15 = K kb*16..kb*16+15
            v8h lo = *(const v8h*)(xr + kt * 32 + kb * 16);
            v8h hi = *(const v8h*)(xr + kt * 32 + kb * 16 + 8);
            v16h a;
#pragma unroll
            for (int i = 0; i < 8; i++) { a[i] = lo[i]; a[i + 8] = hi[i]; }
#pragma unroll
            for (int nt = 0; nt < NT; nt++) {
                const v16h b = *(const v16h*)(wl + ((kt2 * NT + nt) << 9) + (lane << 4));
                acc[nt] = __builtin_amdgcn_wmma_f32_16x16x32_f16(
                    false, a, false, b, (short)0, acc[nt], false, false);
            }
        }
    }

    // Epilogue: lane l holds column n = l&15, rows m = (l>>4)*8 + r
    const int nlo   = lane & 15;
    const int mbase = tok0 + ((lane >> 4) << 3);
#pragma unroll
    for (int nt = 0; nt < NT; nt++) {
        const int   n  = nt * 16 + nlo;
        const float bv = bias[n];
#pragma unroll
        for (int r = 0; r < 8; r++) {
            const int    m   = mbase + r;
            const size_t idx = (size_t)m * NOUT + n;
            float y = acc[nt][r] + bv;
            if (EP == 0)      { Of[idx] = y; }
            else if (EP == 2) { Oh[idx] = (_Float16)geluf(y); }
            else if (EP == 3) { Of[idx] = siluf(y); }
            else if (EP == 4) { Of[idx] = Res[idx] + y; }
            else if (EP == 5) { Acc[idx] += Res[idx] + y; }
        }
    }
}

// ---------------------------------------------------------------------------
// 3x3 conv 64->64 (pad 1) as 9-tap implicit WMMA GEMM over tokens (NHWC f16)
// ---------------------------------------------------------------------------
__global__ __launch_bounds__(256) void conv3x3_wmma(
    const _Float16* __restrict__ X, const _Float16* __restrict__ Wp,
    const float* __restrict__ bias, float* __restrict__ Of,
    _Float16* __restrict__ Oh)
{
    // taps staged 3 at a time: 3 * (KT=2) * (NT=4) * 512 halves = 24KB LDS
    __shared__ __align__(64) _Float16 wl[3 * 8 * 512];

    const int wave = threadIdx.x >> 5;
    const int lane = threadIdx.x & 31;
    const int tok0 = ((blockIdx.x << 3) + wave) << 4;
    const int row  = tok0 + (lane & 15);
    const int kb   = lane >> 4;
    const int b    = row >> 14;
    const int y    = (row >> 7) & 127;
    const int x    = row & 127;

    v8f acc[4];
#pragma unroll
    for (int i = 0; i < 4; i++)
#pragma unroll
        for (int j = 0; j < 8; j++) acc[i][j] = 0.0f;

    for (int tc = 0; tc < 9; tc += 3) {
        __syncthreads();
        async_stage(Wp + (size_t)tc * 8 * 512, wl, 3 * 8 * 1024, threadIdx.x);
        __syncthreads();

        for (int t2 = 0; t2 < 3; t2++) {
            const int tap = tc + t2;
            const int dy  = tap / 3 - 1;
            const int dx  = tap - (tap / 3) * 3 - 1;
            const int yy  = y + dy;
            const int xx  = x + dx;
            const bool valid = ((unsigned)yy < 128u) && ((unsigned)xx < 128u);
            const _Float16* xr = X + ((size_t)(((b << 7) + yy) << 7) + xx) * 64;
#pragma unroll
            for (int kt = 0; kt < 2; kt++) {
                v16h a;
                if (valid) {
                    v8h lo = *(const v8h*)(xr + kt * 32 + kb * 16);
                    v8h hi = *(const v8h*)(xr + kt * 32 + kb * 16 + 8);
#pragma unroll
                    for (int i = 0; i < 8; i++) { a[i] = lo[i]; a[i + 8] = hi[i]; }
                } else {
#pragma unroll
                    for (int i = 0; i < 16; i++) a[i] = (_Float16)0.0f;
                }
#pragma unroll
                for (int nt = 0; nt < 4; nt++) {
                    const v16h bb = *(const v16h*)(wl + (((t2 << 3) + (kt << 2) + nt) << 9) + (lane << 4));
                    acc[nt] = __builtin_amdgcn_wmma_f32_16x16x32_f16(
                        false, a, false, bb, (short)0, acc[nt], false, false);
                }
            }
        }
    }

    const int nlo   = lane & 15;
    const int mbase = tok0 + ((lane >> 4) << 3);
#pragma unroll
    for (int nt = 0; nt < 4; nt++) {
        const int   n  = nt * 16 + nlo;
        const float bv = bias[n];
#pragma unroll
        for (int r = 0; r < 8; r++) {
            const size_t idx = (size_t)(mbase + r) * 64 + n;
            float yv = acc[nt][r] + bv;
            Of[idx] = yv;
            Oh[idx] = (_Float16)yv;
        }
    }
}

// ---------------------------------------------------------------------------
// Haar DWT: NCHW f32 [4,64,256,256] -> A f16 [tok,64] + H-cat f16 [tok,192]
// ---------------------------------------------------------------------------
__global__ void dwt_kernel(const float* __restrict__ in,
                           _Float16* __restrict__ Ah,
                           _Float16* __restrict__ Hc)
{
    int t = blockIdx.x * 256 + threadIdx.x;          // 4*64*128*128 threads
    int c = t & 63, x = (t >> 6) & 127, y = (t >> 13) & 127, b = t >> 20;
    const float* p = in + ((size_t)(b * 64 + c) * 256 + 2 * y) * 256 + 2 * x;
    float p00 = p[0], p01 = p[1], p10 = p[256], p11 = p[257];
    int tok = (b << 14) + (y << 7) + x;
    Ah[(size_t)tok * 64 + c]        = (_Float16)(0.5f * (p00 + p01 + p10 + p11));
    Hc[(size_t)tok * 192 + c]       = (_Float16)(0.5f * (p00 + p01 - p10 - p11));
    Hc[(size_t)tok * 192 + 64 + c]  = (_Float16)(0.5f * (p00 - p01 + p10 - p11));
    Hc[(size_t)tok * 192 + 128 + c] = (_Float16)(0.5f * (p00 - p01 - p10 + p11));
}

// softmax over tokens per (b, channel); in-place on [tok,64] buffer
__global__ __launch_bounds__(256) void softmax_n(float* __restrict__ kh)
{
    __shared__ float red[256];
    int b = blockIdx.x >> 6, c = blockIdx.x & 63;
    size_t base = ((size_t)b << 14) * 64 + c;
    float m = -1e30f;
    for (int n = threadIdx.x; n < 16384; n += 256)
        m = fmaxf(m, kh[base + (size_t)n * 64]);
    red[threadIdx.x] = m; __syncthreads();
    for (int s = 128; s > 0; s >>= 1) {
        if (threadIdx.x < s) red[threadIdx.x] = fmaxf(red[threadIdx.x], red[threadIdx.x + s]);
        __syncthreads();
    }
    m = red[0]; __syncthreads();
    float sum = 0.0f;
    for (int n = threadIdx.x; n < 16384; n += 256)
        sum += expf(kh[base + (size_t)n * 64] - m);
    red[threadIdx.x] = sum; __syncthreads();
    for (int s = 128; s > 0; s >>= 1) {
        if (threadIdx.x < s) red[threadIdx.x] += red[threadIdx.x + s];
        __syncthreads();
    }
    float inv = 1.0f / red[0]; __syncthreads();
    for (int n = threadIdx.x; n < 16384; n += 256) {
        size_t i = base + (size_t)n * 64;
        kh[i] = expf(kh[i] - m) * inv;
    }
}

// kv[b,h,k,v] = sum_n ks[b,n,h*8+k] * vh[b,n,h*8+v]; one block per (b,head)
__global__ __launch_bounds__(256) void kv_kernel(const float* __restrict__ ks,
                                                 const float* __restrict__ vh,
                                                 float* __restrict__ kv)
{
    __shared__ float accs[64];
    int b = blockIdx.x >> 3, h = blockIdx.x & 7;
    if (threadIdx.x < 64) accs[threadIdx.x] = 0.0f;
    __syncthreads();
    float part[64];
#pragma unroll
    for (int i = 0; i < 64; i++) part[i] = 0.0f;
    size_t tb = ((size_t)b << 14);
    for (int n = threadIdx.x; n < 16384; n += 256) {
        size_t off = (tb + n) * 64 + h * 8;
        float kk[8], vv[8];
#pragma unroll
        for (int i = 0; i < 8; i++) { kk[i] = ks[off + i]; vv[i] = vh[off + i]; }
#pragma unroll
        for (int i = 0; i < 8; i++)
#pragma unroll
            for (int j = 0; j < 8; j++) part[i * 8 + j] += kk[i] * vv[j];
    }
    for (int i = 0; i < 64; i++) atomicAdd(&accs[i], part[i]);
    __syncthreads();
    if (threadIdx.x < 64) kv[blockIdx.x * 64 + threadIdx.x] = accs[threadIdx.x];
}

// fa = q @ kv per token-head; x = minus ? fa : vh - fa; store f16
__global__ __launch_bounds__(256) void fa_kernel(const float* __restrict__ qh,
                                                 const float* __restrict__ vh,
                                                 const float* __restrict__ kv,
                                                 _Float16* __restrict__ xh, int minus)
{
    __shared__ float kvs[512];
    int tok = blockIdx.x * 256 + threadIdx.x;
    int b = tok >> 14;
    for (int i = threadIdx.x; i < 512; i += 256) kvs[i] = kv[b * 512 + i];
    __syncthreads();
    const float* q = qh + (size_t)tok * 64;
    const float* v = vh + (size_t)tok * 64;
    _Float16*    o = xh + (size_t)tok * 64;
    for (int h = 0; h < 8; h++) {
        float qq[8];
#pragma unroll
        for (int dk = 0; dk < 8; dk++) qq[dk] = q[h * 8 + dk];
#pragma unroll
        for (int dv = 0; dv < 8; dv++) {
            float s = 0.0f;
#pragma unroll
            for (int dk = 0; dk < 8; dk++) s += qq[dk] * kvs[h * 64 + dk * 8 + dv];
            float xv = minus ? s : (v[h * 8 + dv] - s);
            o[h * 8 + dv] = (_Float16)xv;
        }
    }
}

// LayerNorm over 64 channels per token -> f16
__global__ void ln_kernel(const float* __restrict__ xr, const float* __restrict__ g,
                          const float* __restrict__ bb, _Float16* __restrict__ out)
{
    int tok = blockIdx.x * 256 + threadIdx.x;
    const float* p = xr + (size_t)tok * 64;
    float m = 0.0f;
#pragma unroll
    for (int i = 0; i < 64; i++) m += p[i];
    m *= (1.0f / 64.0f);
    float s2 = 0.0f;
#pragma unroll
    for (int i = 0; i < 64; i++) { float d = p[i] - m; s2 += d * d; }
    float inv = rsqrtf(s2 * (1.0f / 64.0f) + 1e-6f);
    _Float16* o = out + (size_t)tok * 64;
#pragma unroll
    for (int i = 0; i < 64; i++) o[i] = (_Float16)((p[i] - m) * inv * g[i] + bb[i]);
}

__global__ void mul_kernel(float* __restrict__ a, const float* __restrict__ g)
{
    size_t i = (size_t)blockIdx.x * 192 + threadIdx.x;
    a[i] = a[i] * g[i];
}

__global__ __launch_bounds__(256) void bnstats(const float* __restrict__ y, float* __restrict__ st)
{
    __shared__ float r1[256], r2[256];
    int c = blockIdx.x;
    float s = 0.0f, q = 0.0f;
    for (int t = threadIdx.x; t < NTOK; t += 256) {
        float v = y[(size_t)t * 192 + c];
        s += v; q += v * v;
    }
    r1[threadIdx.x] = s; r2[threadIdx.x] = q; __syncthreads();
    for (int k = 128; k > 0; k >>= 1) {
        if (threadIdx.x < k) { r1[threadIdx.x] += r1[threadIdx.x + k]; r2[threadIdx.x] += r2[threadIdx.x + k]; }
        __syncthreads();
    }
    if (threadIdx.x == 0) {
        float mean = r1[0] * (1.0f / NTOK);
        st[c]       = mean;
        st[192 + c] = r2[0] * (1.0f / NTOK) - mean * mean;
    }
}

__global__ void bnapply(float* __restrict__ y, const float* __restrict__ st,
                        const float* __restrict__ g, const float* __restrict__ b)
{
    int c = threadIdx.x;
    size_t i = (size_t)blockIdx.x * 192 + c;
    float v = (y[i] - st[c]) * rsqrtf(st[192 + c] + 1e-5f) * g[c] + b[c];
    y[i] = fmaxf(v, 0.0f);
}

// per-pixel channel max/mean of (gv - gi) into planar [B,2,128,128]
__global__ void cat_kernel(const float* __restrict__ gv, const float* __restrict__ gi,
                           float* __restrict__ cat)
{
    int tok = blockIdx.x * 256 + threadIdx.x;
    int b = tok >> 14, pix = tok & 16383;
    float mx = -1e30f, sm = 0.0f;
    const float* pv = gv + (size_t)tok * 192;
    const float* pi = gi + (size_t)tok * 192;
    for (int c = 0; c < 192; c++) {
        float d = pv[c] - pi[c];
        mx = fmaxf(mx, d); sm += d;
    }
    cat[((size_t)(b * 2)) * 16384 + pix]     = mx;
    cat[((size_t)(b * 2 + 1)) * 16384 + pix] = sm * (1.0f / 192.0f);
}

// 7x7 conv (2->1, pad 3) + sigmoid
__global__ __launch_bounds__(256) void saconv(const float* __restrict__ cat,
                                              const float* __restrict__ w,
                                              float* __restrict__ att)
{
    __shared__ float wlds[98];
    if (threadIdx.x < 98) wlds[threadIdx.x] = w[threadIdx.x];
    __syncthreads();
    int tok = blockIdx.x * 256 + threadIdx.x;
    int b = tok >> 14, y = (tok >> 7) & 127, x = tok & 127;
    float s = 0.0f;
    for (int c = 0; c < 2; c++) {
        const float* pl = cat + ((size_t)(b * 2 + c)) * 16384;
        for (int ky = 0; ky < 7; ky++) {
            int yy = y + ky - 3;
            if ((unsigned)yy >= 128u) continue;
            for (int kx = 0; kx < 7; kx++) {
                int xx = x + kx - 3;
                if ((unsigned)xx >= 128u) continue;
                s += wlds[c * 49 + ky * 7 + kx] * pl[(yy << 7) + xx];
            }
        }
    }
    att[tok] = 1.0f / (1.0f + expf(-s));
}

// F_H = gi + gv * att (in-place into gv)
__global__ void fh_kernel(float* __restrict__ gv, const float* __restrict__ gi,
                          const float* __restrict__ att)
{
    int tok = blockIdx.x;
    size_t i = (size_t)tok * 192 + threadIdx.x;
    gv[i] = gi[i] + gv[i] * att[tok];
}

__global__ void zero_kernel(float* __restrict__ p)
{
    p[(size_t)blockIdx.x * 256 + threadIdx.x] = 0.0f;
}

__global__ void copyhalf_kernel(const float* __restrict__ src, float* __restrict__ dstf,
                                _Float16* __restrict__ dsth)
{
    size_t i = (size_t)blockIdx.x * 256 + threadIdx.x;
    float v = src[i];
    dstf[i] = v;
    dsth[i] = (_Float16)v;
}

// inverse Haar: out NCHW [4,64,256,256]
__global__ void idwt_kernel(const float* __restrict__ FA, const float* __restrict__ FH,
                            float* __restrict__ out)
{
    int t = blockIdx.x * 256 + threadIdx.x;
    int c = t & 63, x = (t >> 6) & 127, y = (t >> 13) & 127, b = t >> 20;
    int tok = (b << 14) + (y << 7) + x;
    float A  = FA[(size_t)tok * 64 + c];
    float H1 = FH[(size_t)tok * 192 + c];
    float H2 = FH[(size_t)tok * 192 + 64 + c];
    float H3 = FH[(size_t)tok * 192 + 128 + c];
    float* o = out + ((size_t)(b * 64 + c) * 256 + 2 * y) * 256 + 2 * x;
    o[0]   = 0.5f * (A + H1 + H2 + H3);
    o[1]   = 0.5f * (A + H1 - H2 - H3);
    o[256] = 0.5f * (A - H1 + H2 - H3);
    o[257] = 0.5f * (A - H1 - H2 + H3);
}

// ---------------------------------------------------------------------------
// Host-side orchestration
// ---------------------------------------------------------------------------
extern "C" void kernel_launch(void* const* d_in, const int* in_sizes, int n_in,
                              void* d_out, int out_size, void* d_ws, size_t ws_size,
                              hipStream_t stream)
{
    (void)in_sizes; (void)n_in; (void)out_size; (void)ws_size;

    // Input order (setup_inputs dict order, depth-first):
    // 0: vi_d   1: ir_d
    // 2..15  cfem1 {wq,bq,wk,bk,wv,bv,wp,bp,ln_g,ln_b,w1,b1,w2,b2}
    // 16..29 cfem2   30..43 dfim1   44..57 dfim2
    // 58 conv1_w  59 conv1_b  60 conv2_w  61 conv2_b
    // 62..67 gate1 {gw,gb,aw,ab,bn_g,bn_b}   68..73 gate2   74 sa_w
    auto F = [&](int i) { return (const float*)d_in[i]; };

    // Workspace carve
    char* wp = (char*)d_ws;
    auto carve = [&](size_t bytes) {
        void* p = (void*)wp;
        wp += (bytes + 255) & ~(size_t)255;
        return p;
    };
    _Float16* xcur_h  = (_Float16*)carve((size_t)NTOK * 64 * 2);  // viA_h, later x/cur
    _Float16* irA_h   = (_Float16*)carve((size_t)NTOK * 64 * 2);
    _Float16* c1_h    = (_Float16*)carve((size_t)NTOK * 64 * 2);
    _Float16* c2_h    = (_Float16*)carve((size_t)NTOK * 64 * 2);
    _Float16* Fc_h    = (_Float16*)carve((size_t)NTOK * 64 * 2);
    _Float16* vhcat_h = (_Float16*)carve((size_t)NTOK * 192 * 2);
    _Float16* ihcat_h = (_Float16*)carve((size_t)NTOK * 192 * 2); // later hid_h
    _Float16* hid_h   = ihcat_h;

    float* c1_f = (float*)carve((size_t)NTOK * 64 * 4);
    float* c2_f = (float*)carve((size_t)NTOK * 64 * 4);
    float* Jreg = (float*)carve((size_t)NTOK * 192 * 4); // gg, later qh|kh|vh
    float* gg   = Jreg;
    float* qh_f = Jreg;
    float* khb  = Jreg + (size_t)NTOK * 64;
    float* vhb  = Jreg + (size_t)NTOK * 128;
    float* ga1  = (float*)carve((size_t)NTOK * 192 * 4); // gate1 -> F_H (lives to end)
    float* ga2  = (float*)carve((size_t)NTOK * 192 * 4); // gate2, later xres|Fc|FA
    float* xres = ga2;
    float* Fc   = ga2 + (size_t)NTOK * 64;
    float* FA   = ga2 + (size_t)NTOK * 128;
    float* catb = (float*)carve((size_t)4 * 2 * 16384 * 4);
    float* attb = (float*)carve((size_t)NTOK * 4);
    float* kvb  = (float*)carve((size_t)2048 * 4);
    float* stat = (float*)carve((size_t)384 * 4);

    // Packed f16 weight fragments (filled once per launch)
    auto carveh = [&](size_t n) { return (_Float16*)carve(n * 2); };
    struct BlkP { _Float16 *wq, *wk, *wv, *wpj, *w1, *w2; } pb[4];
    for (int i = 0; i < 4; i++) {
        pb[i].wq  = carveh(4096);
        pb[i].wk  = carveh(4096);
        pb[i].wv  = carveh(4096);
        pb[i].wpj = carveh(4096);
        pb[i].w1  = carveh(12288);
        pb[i].w2  = carveh(12288);
    }
    _Float16* pg_a[2]; _Float16* pg_g[2];
    pg_a[0] = carveh(36864); pg_g[0] = carveh(36864);
    pg_a[1] = carveh(36864); pg_g[1] = carveh(36864);
    _Float16* pc1 = carveh(36864);
    _Float16* pc2 = carveh(36864);

    const int GB = NTOK / 128;  // 512 blocks for GEMMs
    const int EB = NTOK / 256;  // 256 blocks, thread-per-token kernels
    const int PB = 4 * 64 * HB * HB / 256; // pixel*channel kernels

    // 0) Pre-pack all weights into WMMA fragment order (once per launch)
    auto pp = [&](const float* src, _Float16* dst, int K, int NT) {
        prepack_gemm<<<(K * NT * 16) / 256, 256, 0, stream>>>(src, dst, K, NT);
    };
    const int bases[4] = {2, 16, 30, 44};
    for (int i = 0; i < 4; i++) {
        int bb = bases[i];
        pp(F(bb + 0),  pb[i].wq,  64, 4);
        pp(F(bb + 2),  pb[i].wk,  64, 4);
        pp(F(bb + 4),  pb[i].wv,  64, 4);
        pp(F(bb + 6),  pb[i].wpj, 64, 4);
        pp(F(bb + 10), pb[i].w1,  64, 12);
        pp(F(bb + 12), pb[i].w2, 192, 4);
    }
    pp(F(64), pg_a[0], 192, 12);  // gate1 aw
    pp(F(62), pg_g[0], 192, 12);  // gate1 gw
    pp(F(70), pg_a[1], 192, 12);  // gate2 aw
    pp(F(68), pg_g[1], 192, 12);  // gate2 gw
    prepack_conv<<<144, 256, 0, stream>>>(F(58), pc1);
    prepack_conv<<<144, 256, 0, stream>>>(F(60), pc2);

    // 1) Haar DWT
    dwt_kernel<<<PB, 256, 0, stream>>>(F(0), xcur_h, vhcat_h);
    dwt_kernel<<<PB, 256, 0, stream>>>(F(1), irA_h, ihcat_h);

    // 2) gates (1x1 convs as 192x192 WMMA GEMMs + BN(training) + relu)
    auto run_gate = [&](const _Float16* xin, int gi, int gbase, float* dst) {
        gemm_tok<192, 192, 3><<<GB, 256, 0, stream>>>(xin, pg_a[gi], F(gbase + 3),
                                                      dst, nullptr, nullptr, nullptr);  // a=silu(aw)
        gemm_tok<192, 192, 0><<<GB, 256, 0, stream>>>(xin, pg_g[gi], F(gbase + 1),
                                                      gg, nullptr, nullptr, nullptr);   // g=gw
        mul_kernel<<<NTOK, 192, 0, stream>>>(dst, gg);
        bnstats<<<192, 256, 0, stream>>>(dst, stat);
        bnapply<<<NTOK, 192, 0, stream>>>(dst, stat, F(gbase + 4), F(gbase + 5));
    };
    run_gate(vhcat_h, 0, 62, ga1);
    run_gate(ihcat_h, 1, 68, ga2);

    // 3) spatial fusion: F_H = gi + gv*sigmoid(conv7x7(cat(max,mean))) -> ga1
    cat_kernel<<<EB, 256, 0, stream>>>(ga1, ga2, catb);
    saconv<<<EB, 256, 0, stream>>>(catb, F(74), attb);
    fh_kernel<<<NTOK, 192, 0, stream>>>(ga1, ga2, attb);

    // 4) 3x3 convs on A subbands (implicit WMMA GEMM)
    conv3x3_wmma<<<GB, 256, 0, stream>>>(xcur_h, pc1, F(59), c1_f, c1_h);
    conv3x3_wmma<<<GB, 256, 0, stream>>>(irA_h,  pc2, F(61), c2_f, c2_h);

    // 5) MHCA blocks (all GEMMs on WMMA)
    auto run_mhca = [&](int bi, const _Float16* q_h, const float* q_f,
                        const _Float16* k_h, const _Float16* v_h, int minus, float* dst) {
        int bb = bases[bi];
        gemm_tok<64, 64, 0><<<GB, 256, 0, stream>>>(q_h, pb[bi].wq, F(bb + 1), qh_f, nullptr, nullptr, nullptr);
        gemm_tok<64, 64, 0><<<GB, 256, 0, stream>>>(k_h, pb[bi].wk, F(bb + 3), khb,  nullptr, nullptr, nullptr);
        gemm_tok<64, 64, 0><<<GB, 256, 0, stream>>>(v_h, pb[bi].wv, F(bb + 5), vhb,  nullptr, nullptr, nullptr);
        softmax_n<<<256, 256, 0, stream>>>(khb);                 // ks in-place
        kv_kernel<<<32, 256, 0, stream>>>(khb, vhb, kvb);
        fa_kernel<<<EB, 256, 0, stream>>>(qh_f, vhb, kvb, xcur_h, minus);
        gemm_tok<64, 64, 4><<<GB, 256, 0, stream>>>(xcur_h, pb[bi].wpj, F(bb + 7), xres, nullptr, q_f, nullptr);
        ln_kernel<<<EB, 256, 0, stream>>>(xres, F(bb + 8), F(bb + 9), xcur_h);  // cur reuses x buffer
        gemm_tok<64, 192, 2><<<GB, 256, 0, stream>>>(xcur_h, pb[bi].w1, F(bb + 11), nullptr, hid_h, nullptr, nullptr);
        gemm_tok<192, 64, 5><<<GB, 256, 0, stream>>>(hid_h, pb[bi].w2, F(bb + 13), nullptr, nullptr, xres, dst);
    };

    zero_kernel<<<NTOK * 64 / 256, 256, 0, stream>>>(Fc);
    run_mhca(0, c1_h, c1_f, c2_h, c2_h, 0, Fc);   // cfem1(vi_A, ir_A, ir_A)
    run_mhca(1, c2_h, c2_f, c1_h, c1_h, 0, Fc);   // cfem2(ir_A, vi_A, vi_A)
    copyhalf_kernel<<<NTOK * 64 / 256, 256, 0, stream>>>(Fc, FA, Fc_h);  // FA = Fc, f16 mirror
    run_mhca(2, Fc_h, Fc, c1_h, c1_h, 1, FA);     // dfim1(F_c, vi_A, vi_A)
    run_mhca(3, Fc_h, Fc, c2_h, c2_h, 1, FA);     // dfim2(F_c, ir_A, ir_A)

    // 6) inverse Haar -> output [4,64,256,256] f32
    idwt_kernel<<<PB, 256, 0, stream>>>(FA, ga1, (float*)d_out);
}
